// SANMEncoder_43138651521694
// MI455X (gfx1250) — compile-verified, matrix operand order checked by reading, not run on hardware
//
#include <hip/hip_runtime.h>

// ---------------- problem constants ----------------
static constexpr int Bc  = 4;
static constexpr int Tc  = 1024;
static constexpr int DIN = 560;
static constexpr int Dc  = 512;
static constexpr int Hc  = 4;
static constexpr int FFc = 2048;
static constexpr int Kc  = 11;
static constexpr int Lc  = 6;
static constexpr int DKc = 128;   // D / H

// ---------------- WMMA types ----------------
typedef __attribute__((ext_vector_type(16))) __bf16 v16bf;
typedef __attribute__((ext_vector_type(2)))  __bf16 v2bf;
typedef __attribute__((ext_vector_type(8)))  float  v8f;

union FragBf { v16bf v; unsigned int u[8]; };
struct alignas(16) U4 { unsigned int v[4]; };

__device__ inline unsigned short f2bf(float f) {
  union { __bf16 h; unsigned short s; } x;
  x.h = (__bf16)f;
  return x.s;
}

__device__ inline unsigned int pack2bf(float a, float b) {
#if __has_builtin(__builtin_amdgcn_cvt_pk_bf16_f32)
  union { v2bf v; unsigned int u; } x;
  x.v = __builtin_amdgcn_cvt_pk_bf16_f32(a, b);
  return x.u;
#else
  union { __bf16 h[2]; unsigned int u; } x;
  x.h[0] = (__bf16)a; x.h[1] = (__bf16)b;
  return x.u;
#endif
}

// ---------------------------------------------------------------------------
// WMMA GEMM: C[z] = A[z] (M x K) @ W[z] (K x N)  (+bias +add1 +add2)
// Batch offsets: off = (z/zdiv)*so + (z%zdiv)*si per tensor.
// Block: 256 threads (8 waves), tile 128(M) x 64(N), K-step 64.
// Double-buffered LDS + register pipelining: global loads of tile k+1 overlap
// the 8 WMMAs of tile k; one barrier per K-step.
// REQUIRES: M % 128 == 0, N % 64 == 0 (guaranteed by launch configs).
// Only K may have a tail (K=560 in layer-0 QKV).
// ---------------------------------------------------------------------------
template <bool TRANSB, bool RELU, bool DOCLAMP>
__global__ __launch_bounds__(256)
void gemm_wmma(const float* __restrict__ A, int lda, long soA, long siA,
               const float* __restrict__ W, int ldb, long soB, long siB,
               float* __restrict__ C, int ldc, long soC, long siC,
               const float* __restrict__ bias,
               const float* __restrict__ add1, int ld1, long so1, long si1,
               const float* __restrict__ add2, int ld2, long so2, long si2,
               int M, int N, int K, int zdiv)
{
  __shared__ unsigned short As2[2][128][72];  // [m][k] rows 144B (16B aligned)
  __shared__ unsigned short Bs2[2][64][72];   // [n][k]

  const int tid  = threadIdx.x;
  const int lane = tid & 31;
  const int wid  = tid >> 5;
  const int wm   = wid & 3;    // 4 x 32 rows
  const int wn   = wid >> 2;   // 2 x 32 cols
  const int hi   = lane >> 4;
  const int l16  = lane & 15;

  const int blockN = blockIdx.x * 64;
  const int blockM = blockIdx.y * 128;
  const int z      = blockIdx.z;
  const long zq = z / zdiv, zr = z % zdiv;
  const float* Ab = A + zq * soA + zr * siA;
  const float* Wb = W + zq * soB + zr * siB;
  const long offC = zq * soC + zr * siC;
  const long off1 = zq * so1 + zr * si1;
  const long off2 = zq * so2 + zr * si2;

  // staging coordinates
  const int arow = tid >> 2;         // 0..63 (A row, also TRANSB B row)
  const int akq  = (tid & 3) * 8;    // K sub-offset {0,8,16,24} (+32 2nd half)
  const int bn4  = (tid & 15) * 4;   // !TRANSB: 4 consecutive n
  const int bkk  = tid >> 4;         // !TRANSB: k {0..15} (+16/32/48)

  // -------- tile load (global -> regs) --------
  float4 ra[8];   // A: [(s*2+q)*2 + half]
  float4 rb[4];   // B: TRANSB [(q)*2+half] ; !TRANSB [s]
  auto load_tile = [&](int k0, bool ktail) {
#pragma unroll
    for (int s = 0; s < 2; ++s) {
      const long rowbase = (long)(blockM + arow + s * 64) * lda;
#pragma unroll
      for (int q = 0; q < 2; ++q) {
        const int kq = akq + q * 32;
        const int idx = (s * 2 + q) * 2;
        if (!ktail) {
          ra[idx]     = *(const float4*)(Ab + rowbase + k0 + kq);
          ra[idx + 1] = *(const float4*)(Ab + rowbase + k0 + kq + 4);
        } else {
          float t[8];
#pragma unroll
          for (int j = 0; j < 8; ++j) {
            int ks = k0 + kq + j;
            int ksc = ks < K ? ks : K - 1;
            float v = Ab[rowbase + ksc];
            t[j] = (ks < K) ? v : 0.0f;
          }
          ra[idx]     = make_float4(t[0], t[1], t[2], t[3]);
          ra[idx + 1] = make_float4(t[4], t[5], t[6], t[7]);
        }
      }
    }
    if (TRANSB) {
      const long rowbase = (long)(blockN + arow) * ldb;
#pragma unroll
      for (int q = 0; q < 2; ++q) {
        const int kq = akq + q * 32;
        if (!ktail) {
          rb[q * 2]     = *(const float4*)(Wb + rowbase + k0 + kq);
          rb[q * 2 + 1] = *(const float4*)(Wb + rowbase + k0 + kq + 4);
        } else {
          float t[8];
#pragma unroll
          for (int j = 0; j < 8; ++j) {
            int ks = k0 + kq + j;
            int ksc = ks < K ? ks : K - 1;
            float v = Wb[rowbase + ksc];
            t[j] = (ks < K) ? v : 0.0f;
          }
          rb[q * 2]     = make_float4(t[0], t[1], t[2], t[3]);
          rb[q * 2 + 1] = make_float4(t[4], t[5], t[6], t[7]);
        }
      }
    } else {
#pragma unroll
      for (int s = 0; s < 4; ++s) {
        const int gk = k0 + bkk + s * 16;
        const int gkc = gk < K ? gk : K - 1;
        float4 f = *(const float4*)(Wb + (long)gkc * ldb + blockN + bn4);
        f.x = (gk < K) ? f.x : 0.0f;
        f.y = (gk < K) ? f.y : 0.0f;
        f.z = (gk < K) ? f.z : 0.0f;
        f.w = (gk < K) ? f.w : 0.0f;
        rb[s] = f;
      }
    }
  };

  // -------- tile store (regs -> bf16 LDS) --------
  auto store_tile = [&](int p) {
#pragma unroll
    for (int s = 0; s < 2; ++s) {
      const int r = arow + s * 64;
#pragma unroll
      for (int q = 0; q < 2; ++q) {
        const int idx = (s * 2 + q) * 2;
        U4 pk;
        pk.v[0] = pack2bf(ra[idx].x, ra[idx].y);
        pk.v[1] = pack2bf(ra[idx].z, ra[idx].w);
        pk.v[2] = pack2bf(ra[idx + 1].x, ra[idx + 1].y);
        pk.v[3] = pack2bf(ra[idx + 1].z, ra[idx + 1].w);
        *(U4*)&As2[p][r][akq + q * 32] = pk;
      }
    }
    if (TRANSB) {
#pragma unroll
      for (int q = 0; q < 2; ++q) {
        U4 pk;
        pk.v[0] = pack2bf(rb[q * 2].x, rb[q * 2].y);
        pk.v[1] = pack2bf(rb[q * 2].z, rb[q * 2].w);
        pk.v[2] = pack2bf(rb[q * 2 + 1].x, rb[q * 2 + 1].y);
        pk.v[3] = pack2bf(rb[q * 2 + 1].z, rb[q * 2 + 1].w);
        *(U4*)&Bs2[p][arow][akq + q * 32] = pk;
      }
    } else {
#pragma unroll
      for (int s = 0; s < 4; ++s) {
        const int kk = bkk + s * 16;
        Bs2[p][bn4 + 0][kk] = f2bf(rb[s].x);
        Bs2[p][bn4 + 1][kk] = f2bf(rb[s].y);
        Bs2[p][bn4 + 2][kk] = f2bf(rb[s].z);
        Bs2[p][bn4 + 3][kk] = f2bf(rb[s].w);
      }
    }
  };

  v8f acc00 = {}, acc01 = {}, acc10 = {}, acc11 = {};

  const int nsteps = (K + 63) / 64;
  load_tile(0, 64 > K);
  store_tile(0);
  __syncthreads();

  const int r0 = wm * 32 + l16;
  const int n0 = wn * 32 + l16;

  for (int it = 0; it < nsteps; ++it) {
    const int p = it & 1;
    const bool last = (it == nsteps - 1);

    // issue next-tile global loads before compute (latency overlap)
    if (!last) load_tile((it + 1) * 64, (it + 2) * 64 > K);

    // ---------- compute: 2 k-halves x (2 A frags x 2 B frags) ----------
#pragma unroll
    for (int h2 = 0; h2 < 2; ++h2) {
      const int ko = h2 * 32;
      FragBf fa0, fa1, fb0, fb1;
      *(U4*)&fa0.u[0] = *(const U4*)&As2[p][r0][ko + hi * 8];
      *(U4*)&fa0.u[4] = *(const U4*)&As2[p][r0][ko + 16 + hi * 8];
      *(U4*)&fa1.u[0] = *(const U4*)&As2[p][r0 + 16][ko + hi * 8];
      *(U4*)&fa1.u[4] = *(const U4*)&As2[p][r0 + 16][ko + 16 + hi * 8];
      *(U4*)&fb0.u[0] = *(const U4*)&Bs2[p][n0][ko + hi * 8];
      *(U4*)&fb0.u[4] = *(const U4*)&Bs2[p][n0][ko + 16 + hi * 8];
      *(U4*)&fb1.u[0] = *(const U4*)&Bs2[p][n0 + 16][ko + hi * 8];
      *(U4*)&fb1.u[4] = *(const U4*)&Bs2[p][n0 + 16][ko + 16 + hi * 8];

      acc00 = __builtin_amdgcn_wmma_f32_16x16x32_bf16(false, fa0.v, false, fb0.v,
                                                      (short)0, acc00, false, false);
      acc01 = __builtin_amdgcn_wmma_f32_16x16x32_bf16(false, fa0.v, false, fb1.v,
                                                      (short)0, acc01, false, false);
      acc10 = __builtin_amdgcn_wmma_f32_16x16x32_bf16(false, fa1.v, false, fb0.v,
                                                      (short)0, acc10, false, false);
      acc11 = __builtin_amdgcn_wmma_f32_16x16x32_bf16(false, fa1.v, false, fb1.v,
                                                      (short)0, acc11, false, false);
    }

    // store next tile into the other buffer, then single barrier
    if (!last) store_tile(1 - p);
    __syncthreads();
  }

  // ---------- epilogue: C layout  M = r + 8*hi (per 16-row tile), N = l16 ----
#pragma unroll
  for (int a = 0; a < 2; ++a) {
#pragma unroll
    for (int b = 0; b < 2; ++b) {
      const v8f& acc = (a == 0) ? (b == 0 ? acc00 : acc01)
                                : (b == 0 ? acc10 : acc11);
      const int gn = blockN + wn * 32 + b * 16 + l16;
      const int gmb = blockM + wm * 32 + a * 16 + hi * 8;
      float bv = bias ? bias[gn] : 0.0f;
#pragma unroll
      for (int r = 0; r < 8; ++r) {
        const int gm = gmb + r;
        float v = acc[r] + bv;
        if (add1) v += add1[off1 + (long)gm * ld1 + gn];
        if (add2) v += add2[off2 + (long)gm * ld2 + gn];
        if (RELU)    v = fmaxf(v, 0.0f);
        if (DOCLAMP) v = fminf(fmaxf(v, -60000.0f), 60000.0f);
        C[offC + (long)gm * ldc + gn] = v;
      }
    }
  }
}

// ---------------------------------------------------------------------------
// x * sqrt(D) + sinusoidal position encoding (depth = 560)
// ---------------------------------------------------------------------------
__global__ __launch_bounds__(256)
void embed_kernel(const float* __restrict__ x, float* __restrict__ y, int total)
{
  int i = blockIdx.x * 256 + threadIdx.x;
  if (i >= total) return;
  int d = i % DIN;
  long bt = i / DIN;
  int t = (int)(bt % Tc);
  int dd = (d < DIN / 2) ? d : d - DIN / 2;
  float inv = __expf(-(float)dd * (9.2103403719761836f / 279.0f));
  float ang = (float)(t + 1) * inv;
  float pe  = (d < DIN / 2) ? __sinf(ang) : __cosf(ang);
  y[i] = x[i] * 22.627416997969522f + pe;  // sqrt(512)
}

// ---------------------------------------------------------------------------
// LayerNorm over last dim; one workgroup per row.
// ---------------------------------------------------------------------------
__global__ __launch_bounds__(256)
void ln_kernel(const float* __restrict__ x, const float* __restrict__ g,
               const float* __restrict__ bta, float* __restrict__ y, int cols)
{
  __shared__ float red[256];
  const int row = blockIdx.x, tid = threadIdx.x;
  const float* xr = x + (long)row * cols;
  float* yr = y + (long)row * cols;

  float s = 0.0f;
  for (int i = tid; i < cols; i += 256) s += xr[i];
  red[tid] = s; __syncthreads();
  for (int o = 128; o; o >>= 1) { if (tid < o) red[tid] += red[tid + o]; __syncthreads(); }
  const float mean = red[0] / cols; __syncthreads();

  float s2 = 0.0f;
  for (int i = tid; i < cols; i += 256) { float d = xr[i] - mean; s2 += d * d; }
  red[tid] = s2; __syncthreads();
  for (int o = 128; o; o >>= 1) { if (tid < o) red[tid] += red[tid + o]; __syncthreads(); }
  const float inv = rsqrtf(red[0] / cols + 1e-5f);

  for (int i = tid; i < cols; i += 256)
    yr[i] = (xr[i] - mean) * inv * g[i] + bta[i];
}

// ---------------------------------------------------------------------------
// FSMN memory: mem = (depthwise_conv(v*m, w[D,11], pad 5/5) + v*m) * m
// v lives in fused qkv buffer at column offset 1024, row stride 1536.
// ---------------------------------------------------------------------------
__global__ __launch_bounds__(256)
void fsmn_kernel(const float* __restrict__ qkv, const float* __restrict__ mask,
                 const float* __restrict__ w, float* __restrict__ mem)
{
  int i = blockIdx.x * 256 + threadIdx.x;
  if (i >= Bc * Tc * Dc) return;
  int d = i & (Dc - 1);
  int t = (i >> 9) & (Tc - 1);
  int b = i >> 19;

  float mk = mask[b * Tc + t];
  float vm_t = qkv[((long)(b * Tc + t)) * 1536 + 1024 + d] * mk;

  float acc = 0.0f;
#pragma unroll
  for (int j = 0; j < Kc; ++j) {
    int ts = t + j - 5;
    if (ts >= 0 && ts < Tc) {
      float mk2 = mask[b * Tc + ts];
      float vv = qkv[((long)(b * Tc + ts)) * 1536 + 1024 + d] * mk2;
      acc += w[d * Kc + j] * vv;
    }
  }
  mem[(long)(b * Tc + t) * Dc + d] = (acc + vm_t) * mk;
}

// ---------------------------------------------------------------------------
// Masked, scaled softmax over rows of (B*H*T, T) scores, in place.
// ---------------------------------------------------------------------------
__global__ __launch_bounds__(256)
void softmax_kernel(float* __restrict__ sc, const float* __restrict__ mask, float scale)
{
  __shared__ float red[256];
  const int row = blockIdx.x, tid = threadIdx.x;
  const int b = row >> 12;                 // row / (H*T) = /4096
  float* p = sc + (long)row * Tc;
  const float* mrow = mask + b * Tc;

  float lv[Tc / 256];
  float mx = -1e30f;
#pragma unroll
  for (int i = 0; i < Tc / 256; ++i) {
    int c = tid + i * 256;
    float v = p[c] * scale;
    if (mrow[c] == 0.0f) v = -1e30f;
    lv[i] = v; mx = fmaxf(mx, v);
  }
  red[tid] = mx; __syncthreads();
  for (int o = 128; o; o >>= 1) { if (tid < o) red[tid] = fmaxf(red[tid], red[tid + o]); __syncthreads(); }
  const float MX = red[0]; __syncthreads();

  float sm = 0.0f;
#pragma unroll
  for (int i = 0; i < Tc / 256; ++i) { lv[i] = __expf(lv[i] - MX); sm += lv[i]; }
  red[tid] = sm; __syncthreads();
  for (int o = 128; o; o >>= 1) { if (tid < o) red[tid] += red[tid + o]; __syncthreads(); }
  const float inv = 1.0f / red[0];

#pragma unroll
  for (int i = 0; i < Tc / 256; ++i) {
    int c = tid + i * 256;
    p[c] = (mrow[c] == 0.0f) ? 0.0f : lv[i] * inv;
  }
}

// ---------------------------------------------------------------------------
// host driver
// ---------------------------------------------------------------------------
extern "C" void kernel_launch(void* const* d_in, const int* in_sizes, int n_in,
                              void* d_out, int out_size, void* d_ws, size_t ws_size,
                              hipStream_t stream)
{
  (void)in_sizes; (void)n_in; (void)out_size; (void)ws_size;

  const float* x_in = (const float*)d_in[0];
  const float* mask = (const float*)d_in[1];

  char* ws = (char*)d_ws;
  size_t off = 0;
  auto carve = [&](size_t bytes) -> float* {
    float* p = (float*)(ws + off);
    off = (off + bytes + 255) & ~((size_t)255);
    return p;
  };
  float* xbuf = carve((size_t)Bc * Tc * DIN * 4);
  float* xn   = carve((size_t)Bc * Tc * DIN * 4);
  float* qkvB = carve((size_t)Bc * Tc * 3 * Dc * 4);
  float* mem  = carve((size_t)Bc * Tc * Dc * 4);
  float* sc   = carve((size_t)Bc * Hc * Tc * Tc * 4);
  float* ctx  = carve((size_t)Bc * Tc * Dc * 4);
  float* ffnh = carve((size_t)Bc * Tc * FFc * 4);

  {
    int total = Bc * Tc * DIN;
    embed_kernel<<<(total + 255) / 256, 256, 0, stream>>>(x_in, xbuf, total);
  }

  const int MROWS = Bc * Tc;  // 4096

  for (int l = 0; l < Lc; ++l) {
    const int cols = (l == 0) ? DIN : Dc;
    const float* ln1w  = (l == 0) ? (const float*)d_in[2]  : (const float*)d_in[15] + (l - 1) * Dc;
    const float* ln1b  = (l == 0) ? (const float*)d_in[3]  : (const float*)d_in[16] + (l - 1) * Dc;
    const float* qkvw  = (l == 0) ? (const float*)d_in[4]  : (const float*)d_in[17] + (size_t)(l - 1) * Dc * 3 * Dc;
    const float* qkvb  = (l == 0) ? (const float*)d_in[5]  : (const float*)d_in[18] + (l - 1) * 3 * Dc;
    const float* outw  = (l == 0) ? (const float*)d_in[6]  : (const float*)d_in[19] + (size_t)(l - 1) * Dc * Dc;
    const float* outb  = (l == 0) ? (const float*)d_in[7]  : (const float*)d_in[20] + (l - 1) * Dc;
    const float* fsmnw = (l == 0) ? (const float*)d_in[8]  : (const float*)d_in[21] + (size_t)(l - 1) * Dc * Kc;
    const float* ln2w  = (l == 0) ? (const float*)d_in[9]  : (const float*)d_in[22] + (l - 1) * Dc;
    const float* ln2b  = (l == 0) ? (const float*)d_in[10] : (const float*)d_in[23] + (l - 1) * Dc;
    const float* w1    = (l == 0) ? (const float*)d_in[11] : (const float*)d_in[24] + (size_t)(l - 1) * Dc * FFc;
    const float* b1    = (l == 0) ? (const float*)d_in[12] : (const float*)d_in[25] + (l - 1) * FFc;
    const float* w2    = (l == 0) ? (const float*)d_in[13] : (const float*)d_in[26] + (size_t)(l - 1) * FFc * Dc;
    const float* b2    = (l == 0) ? (const float*)d_in[14] : (const float*)d_in[27] + (l - 1) * Dc;

    // ln1: xbuf -> xn
    ln_kernel<<<MROWS, 256, 0, stream>>>(xbuf, ln1w, ln1b, xn, cols);

    // qkv = xn @ qkv_w + qkv_b : (4096 x cols) @ (cols x 1536)
    {
      dim3 g(3 * Dc / 64, MROWS / 128, 1);
      gemm_wmma<false, false, false><<<g, 256, 0, stream>>>(
          xn, cols, 0, 0, qkvw, 3 * Dc, 0, 0, qkvB, 3 * Dc, 0, 0,
          qkvb, nullptr, 0, 0, 0, nullptr, 0, 0, 0,
          MROWS, 3 * Dc, cols, 1);
    }

    // fsmn memory branch on v
    fsmn_kernel<<<(Bc * Tc * Dc) / 256, 256, 0, stream>>>(qkvB, mask, fsmnw, mem);

    // scores[z] = Q[z] @ K[z]^T, z = b*H + h (16 batches), K=DK=128
    {
      dim3 g(Tc / 64, Tc / 128, Bc * Hc);
      gemm_wmma<true, false, false><<<g, 256, 0, stream>>>(
          qkvB,      3 * Dc, (long)Tc * 3 * Dc, DKc,
          qkvB + Dc, 3 * Dc, (long)Tc * 3 * Dc, DKc,
          sc, Tc, (long)Hc * Tc * Tc, (long)Tc * Tc,
          nullptr, nullptr, 0, 0, 0, nullptr, 0, 0, 0,
          Tc, Tc, DKc, Hc);
    }

    // softmax with 1/sqrt(DK) scale + mask
    softmax_kernel<<<Bc * Hc * Tc, 256, 0, stream>>>(sc, mask, 0.08838834764831845f);

    // ctx[z] = attn[z] @ V[z] : (1024x1024)@(1024x128) -> packed into (B,T,512)
    {
      dim3 g(DKc / 64, Tc / 128, Bc * Hc);
      gemm_wmma<false, false, false><<<g, 256, 0, stream>>>(
          sc, Tc, (long)Hc * Tc * Tc, (long)Tc * Tc,
          qkvB + 2 * Dc, 3 * Dc, (long)Tc * 3 * Dc, DKc,
          ctx, Dc, (long)Tc * Dc, DKc,
          nullptr, nullptr, 0, 0, 0, nullptr, 0, 0, 0,
          Tc, DKc, Tc, Hc);
    }

    // out projection: x = ctx @ out_w + out_b + mem (+ residual for l>0)
    {
      dim3 g(Dc / 64, MROWS / 128, 1);
      gemm_wmma<false, false, false><<<g, 256, 0, stream>>>(
          ctx, Dc, 0, 0, outw, Dc, 0, 0, xbuf, Dc, 0, 0,
          outb, mem, Dc, 0, 0,
          (l > 0) ? xbuf : nullptr, Dc, 0, 0,
          MROWS, Dc, Dc, 1);
    }

    // ln2: xbuf -> xn
    ln_kernel<<<MROWS, 256, 0, stream>>>(xbuf, ln2w, ln2b, xn, Dc);

    // ffn1: ffnh = relu(xn @ w1 + b1)
    {
      dim3 g(FFc / 64, MROWS / 128, 1);
      gemm_wmma<false, true, false><<<g, 256, 0, stream>>>(
          xn, Dc, 0, 0, w1, FFc, 0, 0, ffnh, FFc, 0, 0,
          b1, nullptr, 0, 0, 0, nullptr, 0, 0, 0,
          MROWS, FFc, Dc, 1);
    }

    // ffn2: xbuf = clamp(ffnh @ w2 + b2 + xbuf)
    {
      dim3 g(Dc / 64, MROWS / 128, 1);
      gemm_wmma<false, false, true><<<g, 256, 0, stream>>>(
          ffnh, FFc, 0, 0, w2, Dc, 0, 0, xbuf, Dc, 0, 0,
          b2, xbuf, Dc, 0, 0, nullptr, 0, 0, 0,
          MROWS, Dc, FFc, 1);
    }
  }

  // final layernorm -> d_out
  ln_kernel<<<MROWS, 256, 0, stream>>>(xbuf, (const float*)d_in[28],
                                       (const float*)d_in[29], (float*)d_out, Dc);
}
